// MultiHeadMLP_87943750353236
// MI455X (gfx1250) — compile-verified, matrix operand order checked by reading, not run on hardware
//
#include <hip/hip_runtime.h>
#include <hip/hip_bf16.h>

#define NUM_HEADS 8
#define HEAD_DIM 128
#define NUM_HIDDEN 2048
#define D_MODEL 1024
#define SEQ 4096
#define EPSF 1e-6f

typedef __bf16 bf16_t;
typedef __attribute__((ext_vector_type(16))) __bf16 v16bf;
typedef __attribute__((ext_vector_type(8)))  float  v8f;
typedef __attribute__((ext_vector_type(4)))  int    v4i;

// ---------------------------------------------------------------------------
// WMMA helpers (CDNA5 wave32, 16x16x32 bf16 -> f32 accum)
// ---------------------------------------------------------------------------
static __device__ __forceinline__ v8f wmma_bf16(v16bf a, v16bf b, v8f c) {
  // (neg_a, A, neg_b, B, c_mod, C, reuse_a, reuse_b)
  return __builtin_amdgcn_wmma_f32_16x16x32_bf16(false, a, false, b, (short)0, c,
                                                 false, false);
}

// 16x32 A/B fragment from a row-major [rows x K] bf16 source (global or LDS).
// ISA layout: lane%16 = row; lanes 0-15 hold K in {0..7,16..23},
// lanes 16-31 hold K in {8..15,24..31}; VGPR j holds K pair kb(j), kb(j)+1.
static __device__ __forceinline__ v16bf load_frag_bf16(const bf16_t* base, int ld,
                                                       int k0) {
  const int lane = threadIdx.x & 31;
  const int r = lane & 15;
  const int hf = lane >> 4;
  const bf16_t* p = base + (size_t)r * ld + k0 + hf * 8;
  v16bf f;
#pragma unroll
  for (int j = 0; j < 8; ++j) {
    const int kb = (j < 4) ? (2 * j) : (16 + 2 * (j - 4));
    f[2 * j] = p[kb];
    f[2 * j + 1] = p[kb + 1];
  }
  return f;
}

// Same mapping, converting an f32 source -> bf16 fragment on the fly.
static __device__ __forceinline__ v16bf load_frag_f32(const float* base, int ld,
                                                      int k0) {
  const int lane = threadIdx.x & 31;
  const int r = lane & 15;
  const int hf = lane >> 4;
  const float* p = base + (size_t)r * ld + k0 + hf * 8;
  v16bf f;
#pragma unroll
  for (int j = 0; j < 8; ++j) {
    const int kb = (j < 4) ? (2 * j) : (16 + 2 * (j - 4));
    f[2 * j] = (bf16_t)p[kb];
    f[2 * j + 1] = (bf16_t)p[kb + 1];
  }
  return f;
}

// ---------------------------------------------------------------------------
// Cooperative global->LDS staging: async-DMA path when the toolchain exposes
// the gfx1250 builtin (ASYNCcnt-tracked), else a synchronous b128 bounce.
// Builtin prototype (per compiler diagnostic): (v4i AS1* src, v4i AS3* dst,
// imm offset, imm cpol).
// ---------------------------------------------------------------------------
#if __has_builtin(__builtin_amdgcn_global_load_async_to_lds_b128)
#define HAVE_ASYNC_LDS 1
typedef __attribute__((address_space(1))) v4i* gv4i_p;
typedef __attribute__((address_space(3))) v4i* lv4i_p;
#else
#define HAVE_ASYNC_LDS 0
#endif

static __device__ __forceinline__ void stage_b128(bf16_t* lds_dst,
                                                  const bf16_t* gsrc) {
#if HAVE_ASYNC_LDS
  __builtin_amdgcn_global_load_async_to_lds_b128((gv4i_p)gsrc, (lv4i_p)lds_dst,
                                                 0, 0);
#else
  *(v4i*)lds_dst = *(const v4i*)gsrc;
#endif
}

static __device__ __forceinline__ void wait_stage() {
#if HAVE_ASYNC_LDS
  asm volatile("s_wait_asynccnt 0" ::: "memory");
#else
  asm volatile("s_wait_dscnt 0" ::: "memory");
#endif
}

// ---------------------------------------------------------------------------
// Prep kernels: bf16 operands laid out so every WMMA B-load is a contiguous
// N-major [N][K] read.
// ---------------------------------------------------------------------------

// keys (n,h,d) -> kn[h][n][d] bf16, L2-normalized per (n,h), attn_scale folded.
__global__ void prep_keys_kernel(const float* __restrict__ keys,
                                 const float* __restrict__ attn_scale,
                                 bf16_t* __restrict__ kn) {
  const int wid = (blockIdx.x * blockDim.x + threadIdx.x) >> 5;  // one wave/row
  const int lane = threadIdx.x & 31;
  const int n = wid >> 3;
  const int h = wid & 7;
  const float* row = keys + ((size_t)n * NUM_HEADS + h) * HEAD_DIM;
  float v[4];
  float ss = 0.f;
#pragma unroll
  for (int j = 0; j < 4; ++j) {
    v[j] = row[lane * 4 + j];
    ss += v[j] * v[j];
  }
#pragma unroll
  for (int m = 1; m < 32; m <<= 1) ss += __shfl_xor(ss, m, 32);
  const float sc = rsqrtf(ss + EPSF) * attn_scale[h];
  bf16_t* out = kn + ((size_t)h * NUM_HIDDEN + n) * HEAD_DIM + lane * 4;
#pragma unroll
  for (int j = 0; j < 4; ++j) out[j] = (bf16_t)(v[j] * sc);
}

// values (n,h,d) -> vt[h][d][n] bf16 (so PV B-operand is N-major [d][n]).
__global__ void prep_values_kernel(const float* __restrict__ values,
                                   bf16_t* __restrict__ vt) {
  const int idx = blockIdx.x * blockDim.x + threadIdx.x;
  if (idx >= NUM_HIDDEN * NUM_HEADS * HEAD_DIM) return;
  const int d = idx & (HEAD_DIM - 1);
  const int h = (idx >> 7) & 7;
  const int n = idx >> 10;
  vt[((size_t)h * HEAD_DIM + d) * NUM_HIDDEN + n] = (bf16_t)values[idx];
}

// W[rows][cols] f32 -> WT[cols][rows] bf16
__global__ void prep_transpose_kernel(const float* __restrict__ W,
                                      bf16_t* __restrict__ WT, int rows,
                                      int cols) {
  const int idx = blockIdx.x * blockDim.x + threadIdx.x;
  if (idx >= rows * cols) return;
  const int c = idx % cols;
  const int r = idx / cols;
  WT[(size_t)c * rows + r] = (bf16_t)W[idx];
}

// ---------------------------------------------------------------------------
// Kernel 1: q = x @ Wq, L2-normalize per (s,h) -> qn bf16 [s][h][d]
// One wave = 16 rows x one head (128 cols), K = 1024.
// ---------------------------------------------------------------------------
__global__ __launch_bounds__(256) void qproj_kernel(
    const float* __restrict__ x, const bf16_t* __restrict__ wqt,
    bf16_t* __restrict__ qn) {
  const int wid = (blockIdx.x * blockDim.x + threadIdx.x) >> 5;
  const int lane = threadIdx.x & 31;
  const int r = lane & 15, hf = lane >> 4;
  const int h = wid & 7;
  const int s0 = (wid >> 3) * 16;

  v8f acc[8] = {};
  const float* xbase = x + (size_t)s0 * D_MODEL;
  const bf16_t* wbase = wqt + (size_t)h * HEAD_DIM * D_MODEL;
  for (int k0 = 0; k0 < D_MODEL; k0 += 32) {
    v16bf a = load_frag_f32(xbase, D_MODEL, k0);
#pragma unroll
    for (int f = 0; f < 8; ++f) {
      v16bf b = load_frag_bf16(wbase + (size_t)(f * 16) * D_MODEL, D_MODEL, k0);
      acc[f] = wmma_bf16(a, b, acc[f]);
    }
  }
  // Row-wise sum of squares. C-layout: element i -> row i + 8*hf, one column
  // per lane within each 16-lane half -> xor-reduce over masks 1..8.
  float ss[8];
#pragma unroll
  for (int i = 0; i < 8; ++i) {
    float s = 0.f;
#pragma unroll
    for (int f = 0; f < 8; ++f) s += acc[f][i] * acc[f][i];
    ss[i] = s;
  }
#pragma unroll
  for (int m = 1; m < 16; m <<= 1)
#pragma unroll
    for (int i = 0; i < 8; ++i) ss[i] += __shfl_xor(ss[i], m, 32);
  float inv[8];
#pragma unroll
  for (int i = 0; i < 8; ++i) inv[i] = rsqrtf(ss[i] + EPSF);
#pragma unroll
  for (int f = 0; f < 8; ++f)
#pragma unroll
    for (int i = 0; i < 8; ++i) {
      const int row = s0 + i + hf * 8;
      const int col = f * 16 + r;
      qn[((size_t)row * NUM_HEADS + h) * HEAD_DIM + col] =
          (bf16_t)(acc[f][i] * inv[i]);
    }
}

// ---------------------------------------------------------------------------
// Kernel 2: flash attention against the fixed memory bank.
// One block = one head x 128 query rows (8 waves x 16 rows) so all 8 waves
// share the K/V stream. Each 64-slot chunk (16KB K + 16KB V^T) is staged into
// LDS once per block via async global->LDS, double buffered; online softmax
// keeps the 16x128 output accumulator in registers.
// ---------------------------------------------------------------------------
__global__ __launch_bounds__(256) void attn_kernel(
    const bf16_t* __restrict__ qn, const bf16_t* __restrict__ kn,
    const bf16_t* __restrict__ vt, float* __restrict__ y) {
  __shared__ __align__(16) bf16_t kbuf[2][64 * HEAD_DIM];  // 2 x 16KB
  __shared__ __align__(16) bf16_t vbuf[2][HEAD_DIM * 64];  // 2 x 16KB
  __shared__ __align__(16) bf16_t pbuf[8][16 * 64];        // 16KB
  const int tid = threadIdx.x;
  const int w = tid >> 5;
  const int lane = tid & 31;
  const int r = lane & 15, hf = lane >> 4;
  const int h = blockIdx.x & 7;
  const int s0 = ((blockIdx.x >> 3) * 8 + w) * 16;

  const bf16_t* kbase = kn + (size_t)h * NUM_HIDDEN * HEAD_DIM;
  const bf16_t* vbase = vt + (size_t)h * HEAD_DIM * NUM_HIDDEN;

  // Cooperative stage of one 64-slot chunk into buffer pb (8 b128 per thread).
  auto stage_chunk = [&](int nc, int pb) {
    const bf16_t* gk = kbase + (size_t)nc * HEAD_DIM;  // contiguous 16KB
#pragma unroll
    for (int i = 0; i < 4; ++i) {
      const int off = (tid + i * 256) * 8;  // bf16 elements, 16B granules
      stage_b128(&kbuf[pb][off], gk + off);
    }
#pragma unroll
    for (int i = 0; i < 4; ++i) {
      const int slot = tid + i * 256;  // 1024 b128 granules: 128 rows x 8
      const int row = slot >> 3;
      const int seg = slot & 7;
      stage_b128(&vbuf[pb][row * 64 + seg * 8],
                 vbase + (size_t)row * NUM_HIDDEN + nc + seg * 8);
    }
  };

  // Q fragments: 16 rows x 128 dims, register-resident for the whole pass.
  v16bf qa[4];
  const bf16_t* qbase = qn + ((size_t)s0 * NUM_HEADS + h) * HEAD_DIM;
#pragma unroll
  for (int kk = 0; kk < 4; ++kk)
    qa[kk] = load_frag_bf16(qbase, NUM_HEADS * HEAD_DIM, kk * 32);

  v8f o[8] = {};
  float mrow[8], lrow[8];
#pragma unroll
  for (int i = 0; i < 8; ++i) {
    mrow[i] = -1e30f;
    lrow[i] = 0.f;
  }

  stage_chunk(0, 0);
  int pb = 0;
  for (int nc = 0; nc < NUM_HIDDEN; nc += 64, pb ^= 1) {
    wait_stage();      // this thread's staged granules have landed in LDS
    __syncthreads();   // whole block: chunk ready, prior chunk fully consumed
    if (nc + 64 < NUM_HIDDEN) stage_chunk(nc + 64, pb ^ 1);

    // S tile: 16 rows x 64 slots, QK^T with K-depth 128 (operands from LDS).
    v8f sc[4] = {};
#pragma unroll
    for (int t = 0; t < 4; ++t) {
#pragma unroll
      for (int kk = 0; kk < 4; ++kk) {
        v16bf bf = load_frag_bf16(&kbuf[pb][(t * 16) * HEAD_DIM], HEAD_DIM,
                                  kk * 32);
        sc[t] = wmma_bf16(qa[kk], bf, sc[t]);
      }
    }

    // Online softmax: chunk row-max, rescale running state.
    float cmax[8];
#pragma unroll
    for (int i = 0; i < 8; ++i)
      cmax[i] = fmaxf(fmaxf(sc[0][i], sc[1][i]), fmaxf(sc[2][i], sc[3][i]));
#pragma unroll
    for (int m = 1; m < 16; m <<= 1)
#pragma unroll
      for (int i = 0; i < 8; ++i)
        cmax[i] = fmaxf(cmax[i], __shfl_xor(cmax[i], m, 32));
    float corr[8];
#pragma unroll
    for (int i = 0; i < 8; ++i) {
      const float mn = fmaxf(mrow[i], cmax[i]);
      corr[i] = __expf(mrow[i] - mn);
      mrow[i] = mn;
    }
    float psum[8];
#pragma unroll
    for (int i = 0; i < 8; ++i) {
      float s = 0.f;
#pragma unroll
      for (int t = 0; t < 4; ++t) {
        const float p = __expf(sc[t][i] - mrow[i]);
        sc[t][i] = p;
        s += p;
      }
      psum[i] = s;
    }
#pragma unroll
    for (int m = 1; m < 16; m <<= 1)
#pragma unroll
      for (int i = 0; i < 8; ++i) psum[i] += __shfl_xor(psum[i], m, 32);
#pragma unroll
    for (int i = 0; i < 8; ++i) lrow[i] = lrow[i] * corr[i] + psum[i];
#pragma unroll
    for (int f = 0; f < 8; ++f)
#pragma unroll
      for (int i = 0; i < 8; ++i) o[f][i] *= corr[i];

    // C-layout -> row-major 16x64 bf16 P tile in per-wave LDS scratch.
#pragma unroll
    for (int t = 0; t < 4; ++t)
#pragma unroll
      for (int i = 0; i < 8; ++i)
        pbuf[w][(i + hf * 8) * 64 + t * 16 + r] = (bf16_t)sc[t][i];
    asm volatile("s_wait_dscnt 0" ::: "memory");
    __builtin_amdgcn_wave_barrier();

    // Reload P as A fragments (K = 64 -> two 16x32 frags).
    v16bf pa0 = load_frag_bf16(&pbuf[w][0], 64, 0);
    v16bf pa1 = load_frag_bf16(&pbuf[w][0], 64, 32);
    __builtin_amdgcn_wave_barrier();

    // O += P @ V_chunk (B operand from the staged V^T tile).
#pragma unroll
    for (int f = 0; f < 8; ++f) {
      const bf16_t* vb = &vbuf[pb][(f * 16) * 64];
      v16bf b0 = load_frag_bf16(vb, 64, 0);
      v16bf b1 = load_frag_bf16(vb, 64, 32);
      o[f] = wmma_bf16(pa0, b0, o[f]);
      o[f] = wmma_bf16(pa1, b1, o[f]);
    }
  }

  // Normalize by softmax denominator, emit y f32 [s][h*128+d].
  float inv[8];
#pragma unroll
  for (int i = 0; i < 8; ++i) inv[i] = 1.0f / lrow[i];
#pragma unroll
  for (int f = 0; f < 8; ++f)
#pragma unroll
    for (int i = 0; i < 8; ++i) {
      const int row = s0 + i + hf * 8;
      const int col = h * HEAD_DIM + f * 16 + r;
      y[(size_t)row * (NUM_HEADS * HEAD_DIM) + col] = o[f][i] * inv[i];
    }
}

// ---------------------------------------------------------------------------
// Kernel 3: out = y @ Wo. One wave = 16 rows x 128 output cols, K = 1024.
// ---------------------------------------------------------------------------
__global__ __launch_bounds__(256) void oproj_kernel(
    const float* __restrict__ y, const bf16_t* __restrict__ wot,
    float* __restrict__ out) {
  const int wid = (blockIdx.x * blockDim.x + threadIdx.x) >> 5;
  const int lane = threadIdx.x & 31;
  const int r = lane & 15, hf = lane >> 4;
  const int nt = wid & 7;
  const int s0 = (wid >> 3) * 16;

  v8f acc[8] = {};
  const float* ybase = y + (size_t)s0 * D_MODEL;
  const bf16_t* wbase = wot + (size_t)(nt * 128) * D_MODEL;
  for (int k0 = 0; k0 < D_MODEL; k0 += 32) {
    v16bf a = load_frag_f32(ybase, D_MODEL, k0);
#pragma unroll
    for (int f = 0; f < 8; ++f) {
      v16bf b = load_frag_bf16(wbase + (size_t)(f * 16) * D_MODEL, D_MODEL, k0);
      acc[f] = wmma_bf16(a, b, acc[f]);
    }
  }
#pragma unroll
  for (int f = 0; f < 8; ++f)
#pragma unroll
    for (int i = 0; i < 8; ++i) {
      const int row = s0 + i + hf * 8;
      const int col = nt * 128 + f * 16 + r;
      out[(size_t)row * D_MODEL + col] = acc[f][i];
    }
}

// ---------------------------------------------------------------------------
extern "C" void kernel_launch(void* const* d_in, const int* in_sizes, int n_in,
                              void* d_out, int out_size, void* d_ws,
                              size_t ws_size, hipStream_t stream) {
  (void)in_sizes;
  (void)n_in;
  (void)out_size;
  (void)ws_size;
  const float* x = (const float*)d_in[0];
  const float* Wq = (const float*)d_in[1];
  const float* keys = (const float*)d_in[2];
  const float* values = (const float*)d_in[3];
  const float* attn_scale = (const float*)d_in[4];
  const float* Wo = (const float*)d_in[5];
  float* out = (float*)d_out;

  // Workspace carve (36 MiB total):
  char* ws = (char*)d_ws;
  bf16_t* kn = (bf16_t*)(ws);                   //  4 MiB [h][n][d]
  bf16_t* vt = (bf16_t*)(ws + (4ull << 20));    //  4 MiB [h][d][n]
  bf16_t* wqt = (bf16_t*)(ws + (8ull << 20));   //  2 MiB [n][m]
  bf16_t* wot = (bf16_t*)(ws + (10ull << 20));  //  2 MiB [o][n]
  bf16_t* qn = (bf16_t*)(ws + (12ull << 20));   //  8 MiB [s][h][d]
  float* y = (float*)(ws + (20ull << 20));      // 16 MiB [s][h*d]

  // Prep (memory-bound, ~36 MiB of traffic total).
  prep_keys_kernel<<<(NUM_HIDDEN * NUM_HEADS) / 8, 256, 0, stream>>>(
      keys, attn_scale, kn);
  prep_values_kernel<<<(NUM_HIDDEN * NUM_HEADS * HEAD_DIM) / 256, 256, 0,
                       stream>>>(values, vt);
  prep_transpose_kernel<<<(D_MODEL * D_MODEL) / 256, 256, 0, stream>>>(
      Wq, wqt, D_MODEL, D_MODEL);
  prep_transpose_kernel<<<(D_MODEL * D_MODEL) / 256, 256, 0, stream>>>(
      Wo, wot, D_MODEL, D_MODEL);

  qproj_kernel<<<(SEQ / 16) * NUM_HEADS / 8, 256, 0, stream>>>(x, wqt, qn);
  // attn: 8 heads x 32 groups of 128 query rows = 256 blocks, 8 waves each.
  attn_kernel<<<NUM_HEADS * (SEQ / 128), 256, 0, stream>>>(qn, kn, vt, y);
  oproj_kernel<<<(SEQ / 16) * (D_MODEL / 128) / 8, 256, 0, stream>>>(y, wot,
                                                                     out);
}
